// disVQ_42949673535
// MI455X (gfx1250) — compile-verified, matrix operand order checked by reading, not run on hardware
//
#include <hip/hip_runtime.h>
#include <hip/hip_bf16.h>

// ---------------- problem constants ----------------
#define NUM_CODES 2560
#define DIM       128
#define N_ROWS    65536            // 16*64*64
#define ROWS_PER_BLOCK 128         // 8 waves * 16 rows
#define NBLOCKS   (N_ROWS / ROWS_PER_BLOCK)   // 512
#define CHUNK_CODES 128
#define NCHUNKS   (NUM_CODES / CHUNK_CODES)   // 20
#define TILES_PER_CHUNK (CHUNK_CODES / 16)    // 8
#define TOTAL_ELEMS 8388608.0f

typedef __attribute__((ext_vector_type(16))) _Float16 v16h;
typedef __attribute__((ext_vector_type(8)))  float    v8f;
typedef __attribute__((ext_vector_type(4)))  int      v4i;

typedef __attribute__((address_space(1))) v4i as1_v4i;
typedef __attribute__((address_space(3))) v4i as3_v4i;

union FragH {
  v16h v;
  uint4 q[2];
};

// ---------------- CDNA5 async global->LDS copy (guarded) ----------------
#if defined(__HIP_DEVICE_COMPILE__) && __has_builtin(__builtin_amdgcn_global_load_async_to_lds_b128)
#define HAVE_ASYNC 1
#pragma message("CDNA5 device pass: USING global_load_async_to_lds_b128")
#define ASYNC_CP16(ldsp, gp)                                                 \
  __builtin_amdgcn_global_load_async_to_lds_b128(                            \
      (as1_v4i*)(unsigned long long)(gp),                                    \
      (as3_v4i*)(unsigned)(unsigned long long)(ldsp),                        \
      0, 0)
#else
#define HAVE_ASYNC 0
#if defined(__HIP_DEVICE_COMPILE__)
#pragma message("CDNA5 device pass: async builtin NOT available, using fallback copy")
#endif
#define ASYNC_CP16(ldsp, gp) (*(uint4*)(ldsp) = *(const uint4*)(gp))
#endif

#if HAVE_ASYNC
#if __has_builtin(__builtin_amdgcn_s_wait_asynccnt)
#define WAIT_ASYNC() __builtin_amdgcn_s_wait_asynccnt(0)
#else
#define WAIT_ASYNC() asm volatile("s_wait_asynccnt 0" ::: "memory")
#endif
#else
#define WAIT_ASYNC() ((void)0)
#endif

// ---------------- kernel 1: codebook -> f16 + ||e||^2 ----------------
__global__ void prep_codes(const float* __restrict__ w,
                           _Float16* __restrict__ eh,
                           float* __restrict__ enorm) {
  int j = blockIdx.x * blockDim.x + threadIdx.x;
  if (j >= NUM_CODES) return;
  const float* row = w + (size_t)j * DIM;
  _Float16* dst = eh + (size_t)j * DIM;
  float s = 0.0f;
  for (int k = 0; k < DIM; ++k) {
    float v = row[k];
    s += v * v;
    dst[k] = (_Float16)v;
  }
  enorm[j] = s;
}

// ---------------- kernel 2: [b,c,h,w] f32 -> [n=b*h*w, c] f16 ----------------
__global__ __launch_bounds__(256) void transpose_cast(const float* __restrict__ batch,
                                                      _Float16* __restrict__ zh) {
  __shared__ __align__(16) _Float16 tileLds[64][DIM + 8];
  const int bi   = blockIdx.x >> 6;     // 0..15
  const int tile = blockIdx.x & 63;     // 0..63
  const int hw0  = tile * 64;
  const int tid  = threadIdx.x;

  // read phase: coalesced along hw
  const int hwl = tid & 63;
  const int c0  = tid >> 6;             // 0..3
  const float* src = batch + (size_t)bi * (DIM * 4096) + hw0 + hwl;
  for (int i = 0; i < 32; ++i) {
    int c = c0 + i * 4;
    tileLds[hwl][c] = (_Float16)src[(size_t)c * 4096];
  }
  __syncthreads();

  // write phase: coalesced along c
  const int c  = tid & 127;
  const int r0 = tid >> 7;              // 0..1
  for (int i = 0; i < 32; ++i) {
    int rl = r0 + i * 2;
    zh[(size_t)(bi * 4096 + hw0 + rl) * DIM + c] = tileLds[rl][c];
  }
}

// ---------------- async stage of one 128-code chunk (32 KB) ----------------
__device__ __forceinline__ void stage_chunk(const _Float16* __restrict__ eh,
                                            _Float16* dstLds, int ch, int tid) {
  const uint4* g = (const uint4*)(eh + (size_t)ch * (CHUNK_CODES * DIM));
  uint4* l = (uint4*)dstLds;
#pragma unroll
  for (int i = 0; i < 8; ++i) {
    int idx = i * 256 + tid;            // 2048 x 16B = 32 KB
    ASYNC_CP16(&l[idx], &g[idx]);
  }
}

// ---------------- kernel 3: fused WMMA GEMM + argmin + gather + STE + loss ----------------
__global__ __launch_bounds__(256) void vq_main(const float* __restrict__ batch,
                                               const float* __restrict__ codes_f32,
                                               const _Float16* __restrict__ zh,
                                               const _Float16* __restrict__ eh,
                                               const float* __restrict__ enorm,
                                               float* __restrict__ out,
                                               float* __restrict__ partials) {
  __shared__ __align__(16) _Float16 Bsm[2][CHUNK_CODES * DIM];   // 2 x 32 KB
  __shared__ int   ids[8][16];
  __shared__ float wsum[8];

  const int tid  = threadIdx.x;
  const int wave = tid >> 5;
  const int lane = tid & 31;
  const int rowBase = blockIdx.x * ROWS_PER_BLOCK + wave * 16;

  // ---- A fragments: 16 rows x K=128 f16 in VGPRs ----
  // lane<16 holds M=lane, K sub-blocks {0..7,16..23}; lane>=16 holds K {8..15,24..31}
  const int mrow  = lane & 15;
  const int khalf = (lane >> 4) * 8;    // 0 or 8
  FragH a[4];
  const _Float16* zrow = zh + (size_t)(rowBase + mrow) * DIM;
#pragma unroll
  for (int kk = 0; kk < 4; ++kk) {
    int kb = kk * 32;
    a[kk].q[0] = *(const uint4*)(zrow + kb + khalf);
    a[kk].q[1] = *(const uint4*)(zrow + kb + 16 + khalf);
  }

  float minv[8];
  int   mini[8];
#pragma unroll
  for (int r = 0; r < 8; ++r) { minv[r] = 3.4e38f; mini[r] = 0; }

  stage_chunk(eh, Bsm[0], 0, tid);

  for (int ch = 0; ch < NCHUNKS; ++ch) {
    WAIT_ASYNC();
    __syncthreads();                       // chunk `ch` resident for all waves
    if (ch + 1 < NCHUNKS) stage_chunk(eh, Bsm[(ch + 1) & 1], ch + 1, tid);

    // preload the 8 per-tile code norms for this chunk (hides global latency)
    float en[TILES_PER_CHUNK];
#pragma unroll
    for (int t = 0; t < TILES_PER_CHUNK; ++t)
      en[t] = enorm[ch * CHUNK_CODES + t * 16 + mrow];

    const _Float16* bsm = Bsm[ch & 1];
#pragma unroll
    for (int t = 0; t < TILES_PER_CHUNK; ++t) {
      const _Float16* brow = bsm + (size_t)(t * 16 + mrow) * DIM;  // code column N=mrow

      // preload ALL 4 B fragments -> 8 ds_load_b128 in flight, single wait
      FragH b[4];
#pragma unroll
      for (int kk = 0; kk < 4; ++kk) {
        int kb = kk * 32;
        b[kk].q[0] = *(const uint4*)(brow + kb + khalf);
        b[kk].q[1] = *(const uint4*)(brow + kb + 16 + khalf);
      }

      // two independent accumulator chains to shorten the C-RAW chain
      v8f acc0 = {};
      v8f acc1 = {};
      acc0 = __builtin_amdgcn_wmma_f32_16x16x32_f16(false, a[0].v, false, b[0].v,
                                                    (short)0, acc0, false, false);
      acc1 = __builtin_amdgcn_wmma_f32_16x16x32_f16(false, a[1].v, false, b[1].v,
                                                    (short)0, acc1, false, false);
      acc0 = __builtin_amdgcn_wmma_f32_16x16x32_f16(false, a[2].v, false, b[2].v,
                                                    (short)0, acc0, false, false);
      acc1 = __builtin_amdgcn_wmma_f32_16x16x32_f16(false, a[3].v, false, b[3].v,
                                                    (short)0, acc1, false, false);

      const int j = ch * CHUNK_CODES + t * 16 + mrow;   // global code index in this lane
      const float e2 = en[t];
#pragma unroll
      for (int r = 0; r < 8; ++r) {
        float s = e2 - 2.0f * (acc0[r] + acc1[r]);  // ||e||^2 - 2 z.e
        if (s < minv[r]) { minv[r] = s; mini[r] = j; }
      }
    }
  }

  // ---- argmin butterfly over the 16 lanes of each half (tie -> lower index) ----
#pragma unroll
  for (int r = 0; r < 8; ++r) {
#pragma unroll
    for (int m = 8; m >= 1; m >>= 1) {
      float ov = __shfl_xor(minv[r], m, 32);
      int   oi = __shfl_xor(mini[r], m, 32);
      if (ov < minv[r] || (ov == minv[r] && oi < mini[r])) { minv[r] = ov; mini[r] = oi; }
    }
  }
  if (lane == 0) {
#pragma unroll
    for (int r = 0; r < 8; ++r) ids[wave][r] = mini[r];       // rows 0..7
  }
  if (lane == 16) {
#pragma unroll
    for (int r = 0; r < 8; ++r) ids[wave][8 + r] = mini[r];   // rows 8..15
  }
  __syncthreads();

  // ---- gather + straight-through output + loss partial ----
  float lsum = 0.0f;
  const int m    = lane & 15;
  const int row  = rowBase + m;          // flat n = (b*64+h)*64+w
  const int bi   = row >> 12;            // /4096
  const int hw   = row & 4095;
  const int code = ids[wave][m];
  const float* crow = codes_f32 + (size_t)code * DIM;
  const size_t base = (size_t)bi * (DIM * 4096) + hw;

#pragma unroll 8
  for (int c2 = 0; c2 < 64; ++c2) {
    int c = 2 * c2 + (lane >> 4);
    size_t addr = base + (size_t)c * 4096;     // [b,c,h,w] layout
    float zq = crow[c];
    float z  = batch[addr];
    float d  = zq - z;                         // z_q - z
    float o  = z + d;                          // STE forward value
    float d3 = z - o;                          // batch - out
    out[addr] = o;
    lsum += 2.0f * d * d + 50.0f * d3 * d3;
  }

  // wave reduce, then block reduce into deterministic per-block partial
#pragma unroll
  for (int mm = 16; mm >= 1; mm >>= 1) lsum += __shfl_xor(lsum, mm, 32);
  if (lane == 0) wsum[wave] = lsum;
  __syncthreads();
  if (tid == 0) {
    float s = 0.0f;
    for (int i = 0; i < 8; ++i) s += wsum[i];
    partials[blockIdx.x] = s;
  }
}

// ---------------- kernel 4: deterministic final loss ----------------
__global__ void finalize(const float* __restrict__ partials, float* __restrict__ loss) {
  if (threadIdx.x == 0 && blockIdx.x == 0) {
    float s = 0.0f;
    for (int i = 0; i < NBLOCKS; ++i) s += partials[i];
    *loss = s / TOTAL_ELEMS;
  }
}

// ---------------- launcher ----------------
extern "C" void kernel_launch(void* const* d_in, const int* in_sizes, int n_in,
                              void* d_out, int out_size, void* d_ws, size_t ws_size,
                              hipStream_t stream) {
  const float* batch = (const float*)d_in[0];   // [16,128,64,64] f32
  const float* vqw   = (const float*)d_in[1];   // [2560,128] f32
  float* out = (float*)d_out;                   // 8388608 floats + 1 loss

  char* ws = (char*)d_ws;
  _Float16* zh       = (_Float16*)(ws);                               // 16 MB
  _Float16* eh       = (_Float16*)(ws + 16777216);                    // 640 KB
  float*    enorm    = (float*)  (ws + 16777216 + 655360);            // 10 KB
  float*    partials = (float*)  (ws + 16777216 + 655360 + 10240);    // 2 KB

  prep_codes    <<<(NUM_CODES + 255) / 256, 256, 0, stream>>>(vqw, eh, enorm);
  transpose_cast<<<16 * 64, 256, 0, stream>>>(batch, zh);
  vq_main       <<<NBLOCKS, 256, 0, stream>>>(batch, vqw, zh, eh, enorm, out, partials);
  finalize      <<<1, 32, 0, stream>>>(partials, out + 8388608);
}
